// SkipCausalAttention_10746008175036
// MI455X (gfx1250) — matrix operand order repaired and checked by reading, not compiled, not run on hardware
//
#include <hip/hip_runtime.h>

// ---------------------------------------------------------------------------
// Problem constants (match reference setup_inputs)
// ---------------------------------------------------------------------------
#define BB    2
#define NT    64
#define NC    1024
#define NN    1024
#define NTOK  (NT + NC + NN)   // 2112
#define CDIM  1024
#define HH    16
#define HD    64
#define SCALE 0.125f           // 1/sqrt(64)
#define NEGF  (-1.0e30f)

typedef __attribute__((ext_vector_type(16))) __bf16 bf16x16;
typedef __attribute__((ext_vector_type(8)))  float  floatx8;

struct __align__(16) U4 { unsigned int w[4]; };

union FragA { bf16x16 v; unsigned int u[8]; };
union FragB { bf16x16 v; U4 q[2]; };

// f32 -> bf16 (round to nearest even)
__device__ __forceinline__ unsigned short f2bf(float x) {
  union { float f; unsigned int u; } c; c.f = x;
  unsigned int u = c.u;
  unsigned int r = u + 0x7fffu + ((u >> 16) & 1u);
  return (unsigned short)(r >> 16);
}

// A-matrix fragment (16 rows x 32 K, bf16). rowk0 points at this lane's row,
// at the current K chunk. kbase = 8*(lane>>4). Elements at k = 2*kbase + s
// (contiguous 16 K per lane half, matching ld_frag_b).
__device__ __forceinline__ bf16x16 ld_frag_a(const unsigned short* rowk0, int kbase) {
  FragA f;
#pragma unroll
  for (int v = 0; v < 8; ++v) {
    int k = 2 * kbase + (v << 1);
    f.u[v] = *(const unsigned int*)(rowk0 + k);
  }
  return f.v;
}

// B-matrix fragment (32 K x 16 cols, bf16). rowk0 points at the memory row
// holding this lane's B-column (contiguous over K). kbase2 = 16*(lane>>4).
__device__ __forceinline__ bf16x16 ld_frag_b(const unsigned short* rowk0, int kbase2) {
  FragB f;
  const U4* p = (const U4*)(rowk0 + kbase2);
  f.q[0] = p[0];
  f.q[1] = p[1];
  return f.v;
}

__device__ __forceinline__ floatx8 wmma_bf16(bf16x16 a, bf16x16 b, floatx8 c) {
  return __builtin_amdgcn_wmma_f32_16x16x32_bf16(false, a, false, b, (short)0, c,
                                                 false, false);
}

__device__ __forceinline__ float wsum32(float v) {
#pragma unroll
  for (int m = 16; m > 0; m >>= 1) v += __shfl_xor(v, m, 32);
  return v;
}
__device__ __forceinline__ float wsum16(float v) {
#pragma unroll
  for (int m = 8; m > 0; m >>= 1) v += __shfl_xor(v, m, 32);
  return v;
}
__device__ __forceinline__ float wmax16(float v) {
#pragma unroll
  for (int m = 8; m > 0; m >>= 1) v = fmaxf(v, __shfl_xor(v, m, 32));
  return v;
}

// ---------------------------------------------------------------------------
// f32 -> bf16 bulk convert
// ---------------------------------------------------------------------------
__global__ __launch_bounds__(256) void cvt_bf16_kernel(const float* __restrict__ in,
                                                       unsigned short* __restrict__ out,
                                                       int n) {
  int i = blockIdx.x * 256 + threadIdx.x;
  if (i < n) out[i] = f2bf(in[i]);
}

// ---------------------------------------------------------------------------
// GEMM: C[b, m, n] = sum_k A[b, m, k] * W[n, k] + bias[n]
// A bf16 [*, K] (batch stride strideAb elems), W bf16 [Nout, K], C f32.
// Block = 256 thr (8 waves, 2x4): block tile 64(M) x 256(N);
// each wave: 32x64 = 8 WMMAs per 32-K chunk for 6 fragment loads.
// Requires M % 64 == 0, Nout % 256 == 0, K % 32 == 0 (true for all uses).
// ---------------------------------------------------------------------------
__global__ __launch_bounds__(256) void gemm_bf16_kernel(
    const unsigned short* __restrict__ A, long strideAb,
    const unsigned short* __restrict__ W,
    const float* __restrict__ bias,
    float* __restrict__ C, long strideCb,
    int Nout, int K, int ldc) {
  const int b    = blockIdx.z;
  const int tid  = threadIdx.x;
  const int wid  = tid >> 5;
  const int lane = tid & 31;
  const int wm   = wid >> 2;       // 0..1
  const int wn   = wid & 3;        // 0..3
  const int cn   = lane & 15;
  const int half = lane >> 4;
  const int kba  = half * 8;       // A gather base
  const int kbb  = half * 16;      // B gather base

  const int row0 = blockIdx.y * 64 + wm * 32;
  const int col0 = blockIdx.x * 256 + wn * 64;

  const unsigned short* Ab = A + (size_t)b * strideAb;
  float* Cb = C + (size_t)b * strideCb;

  const unsigned short* arow0 = Ab + (size_t)(row0 + cn) * K;
  const unsigned short* arow1 = Ab + (size_t)(row0 + 16 + cn) * K;
  const unsigned short* wrow[4];
#pragma unroll
  for (int j = 0; j < 4; ++j) wrow[j] = W + (size_t)(col0 + j * 16 + cn) * K;

  floatx8 acc0[4] = {{}, {}, {}, {}};
  floatx8 acc1[4] = {{}, {}, {}, {}};

  for (int k0 = 0; k0 < K; k0 += 32) {
    bf16x16 a0 = ld_frag_a(arow0 + k0, kba);
    bf16x16 a1 = ld_frag_a(arow1 + k0, kba);
#pragma unroll
    for (int j = 0; j < 4; ++j) {
      bf16x16 bj = ld_frag_b(wrow[j] + k0, kbb);
      acc0[j] = wmma_bf16(a0, bj, acc0[j]);
      acc1[j] = wmma_bf16(a1, bj, acc1[j]);
    }
  }

#pragma unroll
  for (int j = 0; j < 4; ++j) {
    const float bia = bias[col0 + j * 16 + cn];
#pragma unroll
    for (int r = 0; r < 8; ++r) {
      int mr = r + 8 * half;
      Cb[(size_t)(row0 + mr) * ldc + col0 + j * 16 + cn]      = acc0[j][r] + bia;
      Cb[(size_t)(row0 + 16 + mr) * ldc + col0 + j * 16 + cn] = acc1[j][r] + bia;
    }
  }
}

// ---------------------------------------------------------------------------
// LN (per-head, over HD=64) + RoPE + layout to:
//   Qh, Kh : [B, H, NTOK, 64] bf16   (row-major over d -> A/B frag friendly)
//   Vt     : [B, H, 64, NTOK] bf16   (transposed -> contiguous-K B frags)
// Block = 128 thr (4 waves); one wave per (b, n, h) unit, 2 dims per lane.
// ---------------------------------------------------------------------------
__global__ __launch_bounds__(128) void lnrope_kernel(
    const float* __restrict__ qkv,       // [B, NTOK, 3072]
    const int* __restrict__ pos_img,     // [B, NC+NN]
    const int* __restrict__ pos_txt,     // [B, NT]
    const float* __restrict__ qn_w, const float* __restrict__ qn_b,
    const float* __restrict__ kn_w, const float* __restrict__ kn_b,
    unsigned short* __restrict__ Qh,
    unsigned short* __restrict__ Kh,
    unsigned short* __restrict__ Vt) {
  const int unit = blockIdx.x * 4 + (threadIdx.x >> 5);
  if (unit >= BB * NTOK * HH) return;
  const int lane = threadIdx.x & 31;

  const int h = unit % HH;
  const int n = (unit / HH) % NTOK;
  const int b = unit / (HH * NTOK);

  const int seg = (n < NT) ? 0 : ((n < NT + NC) ? 1 : 2);
  const int pos = (seg == 0) ? pos_txt[b * NT + n]
                             : pos_img[b * (NC + NN) + (n - NT)];

  const int d0 = lane, d1 = lane + 32;
  const size_t base = (size_t)(b * NTOK + n) * (3 * CDIM) + h * HD;

  float q0 = qkv[base + d0],            q1 = qkv[base + d1];
  float k0 = qkv[base + CDIM + d0],     k1 = qkv[base + CDIM + d1];
  float v0 = qkv[base + 2 * CDIM + d0], v1 = qkv[base + 2 * CDIM + d1];

  // LayerNorm over 64 dims
  const float muq = wsum32(q0 + q1) * (1.0f / 64.0f);
  const float muk = wsum32(k0 + k1) * (1.0f / 64.0f);
  float dq0 = q0 - muq, dq1 = q1 - muq;
  float dk0 = k0 - muk, dk1 = k1 - muk;
  const float vq = wsum32(dq0 * dq0 + dq1 * dq1) * (1.0f / 64.0f);
  const float vk = wsum32(dk0 * dk0 + dk1 * dk1) * (1.0f / 64.0f);
  const float rq = rsqrtf(vq + 1e-5f), rk = rsqrtf(vk + 1e-5f);
  q0 = dq0 * rq * qn_w[seg * HD + d0] + qn_b[seg * HD + d0];
  q1 = dq1 * rq * qn_w[seg * HD + d1] + qn_b[seg * HD + d1];
  k0 = dk0 * rk * kn_w[seg * HD + d0] + kn_b[seg * HD + d0];
  k1 = dk1 * rk * kn_w[seg * HD + d1] + kn_b[seg * HD + d1];

  // RoPE (rotate-half, half = 32): lane owns pair (d0, d0+32)
  const float inv = __powf(10000.0f, -(float)lane * (1.0f / 32.0f));
  const float ang = (float)pos * inv;
  const float cs = __cosf(ang), sn = __sinf(ang);
  const float qr0 = q0 * cs - q1 * sn, qr1 = q1 * cs + q0 * sn;
  const float kr0 = k0 * cs - k1 * sn, kr1 = k1 * cs + k0 * sn;

  const size_t qkbase = ((size_t)(b * HH + h) * NTOK + n) * HD;
  Qh[qkbase + d0] = f2bf(qr0);
  Qh[qkbase + d1] = f2bf(qr1);
  Kh[qkbase + d0] = f2bf(kr0);
  Kh[qkbase + d1] = f2bf(kr1);

  const size_t vtb = (size_t)(b * HH + h) * HD;
  Vt[(vtb + d0) * NTOK + n] = f2bf(v0);
  Vt[(vtb + d1) * NTOK + n] = f2bf(v1);
}

// ---------------------------------------------------------------------------
// Flash attention, causal. One wave per (b, h, 32-row q tile); key tiles of 32.
// Per key tile: 8 S-WMMAs + 8 PV-WMMAs against 4 K-frag + 4 V-frag loads.
// Online softmax in f32; P goes through a 2KB LDS round-trip to reshape
// D-layout f32 -> A-layout bf16. Output bf16 [B, NTOK, C] for the proj GEMM.
// ---------------------------------------------------------------------------
__global__ __launch_bounds__(32) void flash_attn_kernel(
    const unsigned short* __restrict__ Qh,
    const unsigned short* __restrict__ Kh,
    const unsigned short* __restrict__ Vt,
    unsigned short* __restrict__ Obf) {
  const int qt = blockIdx.x, h = blockIdx.y, b = blockIdx.z;
  const int qi0  = qt * 32;
  const int lane = threadIdx.x;
  const int cn   = lane & 15;
  const int half = lane >> 4;
  const int kba  = half * 8;
  const int kbb  = half * 16;

  __shared__ unsigned short Pl[32 * 32];

  const unsigned short* qbase = Qh + (size_t)(b * HH + h) * NTOK * HD;
  const unsigned short* kbase = Kh + (size_t)(b * HH + h) * NTOK * HD;
  const unsigned short* vbase = Vt + (size_t)(b * HH + h) * HD * NTOK;

  // Q fragments for the two 16-row tiles, two 32-K chunks each
  bf16x16 qa[2][2];
#pragma unroll
  for (int t = 0; t < 2; ++t) {
    const unsigned short* qrow = qbase + (size_t)(qi0 + t * 16 + cn) * HD;
    qa[t][0] = ld_frag_a(qrow, kba);
    qa[t][1] = ld_frag_a(qrow + 32, kba);
  }

  floatx8 o[2][4] = {{{}, {}, {}, {}}, {{}, {}, {}, {}}};
  float m[2][8], l[2][8];
#pragma unroll
  for (int t = 0; t < 2; ++t)
#pragma unroll
    for (int r = 0; r < 8; ++r) { m[t][r] = NEGF; l[t][r] = 0.0f; }

  const int nkt = qi0 / 32 + 1;
  for (int kt = 0; kt < nkt; ++kt) {
    const int kj0 = kt * 32;

    // K fragments: two 16-key subtiles x two 32-K chunks
    bf16x16 kb[2][2];
#pragma unroll
    for (int j = 0; j < 2; ++j) {
      const unsigned short* kr = kbase + (size_t)(kj0 + j * 16 + cn) * HD;
      kb[j][0] = ld_frag_b(kr, kbb);
      kb[j][1] = ld_frag_b(kr + 32, kbb);
    }

#pragma unroll
    for (int t = 0; t < 2; ++t) {
      floatx8 s[2] = {{}, {}};
#pragma unroll
      for (int j = 0; j < 2; ++j) {
        s[j] = wmma_bf16(qa[t][0], kb[j][0], s[j]);
        s[j] = wmma_bf16(qa[t][1], kb[j][1], s[j]);
      }
#pragma unroll
      for (int r = 0; r < 8; ++r) {
        const int qrg = qi0 + t * 16 + r + 8 * half;
        float v0 = s[0][r] * SCALE; if (kj0 + cn > qrg)      v0 = NEGF;
        float v1 = s[1][r] * SCALE; if (kj0 + 16 + cn > qrg) v1 = NEGF;
        const float mx   = wmax16(fmaxf(v0, v1));
        const float mnew = fmaxf(m[t][r], mx);
        const float corr = __expf(m[t][r] - mnew);
        const float p0 = __expf(v0 - mnew);
        const float p1 = __expf(v1 - mnew);
        l[t][r] = l[t][r] * corr + wsum16(p0 + p1);
        m[t][r] = mnew;
        o[t][0][r] *= corr; o[t][1][r] *= corr;
        o[t][2][r] *= corr; o[t][3][r] *= corr;
        Pl[(t * 16 + r + 8 * half) * 32 + cn]      = f2bf(p0);
        Pl[(t * 16 + r + 8 * half) * 32 + 16 + cn] = f2bf(p1);
      }
    }
    __syncthreads();

    // P A-fragments (rows 0-15 and 16-31 of the 32x32 P tile)
    const bf16x16 pa0 = ld_frag_a(&Pl[cn * 32], kba);
    const bf16x16 pa1 = ld_frag_a(&Pl[(16 + cn) * 32], kba);
#pragma unroll
    for (int vt = 0; vt < 4; ++vt) {
      const unsigned short* vr =
          vbase + (size_t)(vt * 16 + cn) * NTOK + kj0;
      const bf16x16 vb = ld_frag_b(vr, kbb);
      o[0][vt] = wmma_bf16(pa0, vb, o[0][vt]);
      o[1][vt] = wmma_bf16(pa1, vb, o[1][vt]);
    }
  }

#pragma unroll
  for (int t = 0; t < 2; ++t)
#pragma unroll
    for (int r = 0; r < 8; ++r) {
      const float invl = 1.0f / l[t][r];
      const int n = qi0 + t * 16 + r + 8 * half;
      unsigned short* orow = Obf + (size_t)(b * NTOK + n) * CDIM + h * HD;
      orow[0 * 16 + cn] = f2bf(o[t][0][r] * invl);
      orow[1 * 16 + cn] = f2bf(o[t][1][r] * invl);
      orow[2 * 16 + cn] = f2bf(o[t][2][r] * invl);
      orow[3 * 16 + cn] = f2bf(o[t][3][r] * invl);
    }
}

// ---------------------------------------------------------------------------
// Host-side launch
// ---------------------------------------------------------------------------
extern "C" void kernel_launch(void* const* d_in, const int* in_sizes, int n_in,
                              void* d_out, int out_size, void* d_ws, size_t ws_size,
                              hipStream_t stream) {
  (void)in_sizes; (void)n_in; (void)out_size; (void)ws_size;

  const float* x        = (const float*)d_in[0];
  const int*   pos_img  = (const int*)d_in[1];
  const int*   pos_txt  = (const int*)d_in[2];
  const float* qkv_w    = (const float*)d_in[7];
  const float* qkv_b    = (const float*)d_in[8];
  const float* qn_w     = (const float*)d_in[9];
  const float* qn_b     = (const float*)d_in[10];
  const float* kn_w     = (const float*)d_in[11];
  const float* kn_b     = (const float*)d_in[12];
  const float* proj_w   = (const float*)d_in[13];
  const float* proj_b   = (const float*)d_in[14];
  float* out = (float*)d_out;

  // Workspace carve
  size_t off = 0;
  auto carve = [&](size_t bytes) {
    void* p = (char*)d_ws + off;
    off += (bytes + 255) & ~(size_t)255;
    return p;
  };
  const size_t nX  = (size_t)BB * NTOK * CDIM;           // 4,325,376
  const size_t nWq = (size_t)3 * 3 * CDIM * CDIM;        // 9,437,184
  const size_t nWp = (size_t)3 * CDIM * CDIM;            // 3,145,728
  unsigned short* x_bf    = (unsigned short*)carve(nX * 2);
  unsigned short* wqkv_bf = (unsigned short*)carve(nWq * 2);
  unsigned short* wprj_bf = (unsigned short*)carve(nWp * 2);
  float*          qkv_f   = (float*)carve((size_t)BB * NTOK * 3 * CDIM * 4);
  unsigned short* Qh      = (unsigned short*)carve(nX * 2);
  unsigned short* Kh      = (unsigned short*)carve(nX * 2);
  unsigned short* Vt      = (unsigned short*)carve(nX * 2);
  unsigned short* attn_bf = (unsigned short*)carve(nX * 2);

  // 1) bf16 conversions
  cvt_bf16_kernel<<<dim3((unsigned)((nX  + 255) / 256)), 256, 0, stream>>>(x,      x_bf,    (int)nX);
  cvt_bf16_kernel<<<dim3((unsigned)((nWq + 255) / 256)), 256, 0, stream>>>(qkv_w,  wqkv_bf, (int)nWq);
  cvt_bf16_kernel<<<dim3((unsigned)((nWp + 255) / 256)), 256, 0, stream>>>(proj_w, wprj_bf, (int)nWp);

  const int segS[3]  = {NT, NC, NN};
  const int segT0[3] = {0, NT, NT + NC};

  // 2) Per-segment QKV projection: [S,1024] x [3072,1024]^T -> [S,3072]
  for (int s = 0; s < 3; ++s) {
    dim3 grid(3 * CDIM / 256, segS[s] / 64, BB);
    gemm_bf16_kernel<<<grid, 256, 0, stream>>>(
        x_bf + (size_t)segT0[s] * CDIM, (long)NTOK * CDIM,
        wqkv_bf + (size_t)s * 3 * CDIM * CDIM,
        qkv_b + (size_t)s * 3 * CDIM,
        qkv_f + (size_t)segT0[s] * 3 * CDIM, (long)NTOK * 3 * CDIM,
        3 * CDIM, CDIM, 3 * CDIM);
  }

  // 3) LN + RoPE + head-major / transposed-V layouts
  {
    const int units = BB * NTOK * HH;
    lnrope_kernel<<<dim3((units + 3) / 4), 128, 0, stream>>>(
        qkv_f, pos_img, pos_txt, qn_w, qn_b, kn_w, kn_b, Qh, Kh, Vt);
  }

  // 4) Flash attention (causal), 32 query rows per wave
  flash_attn_kernel<<<dim3(NTOK / 32, HH, BB), 32, 0, stream>>>(Qh, Kh, Vt, attn_bf);

  // 5) Per-segment output projection -> d_out f32 [B, NTOK, C]
  for (int s = 0; s < 3; ++s) {
    dim3 grid(CDIM / 256, segS[s] / 64, BB);
    gemm_bf16_kernel<<<grid, 256, 0, stream>>>(
        attn_bf + (size_t)segT0[s] * CDIM, (long)NTOK * CDIM,
        wprj_bf + (size_t)s * CDIM * CDIM,
        proj_b + (size_t)s * CDIM,
        out + (size_t)segT0[s] * CDIM, (long)NTOK * CDIM,
        CDIM, CDIM, CDIM);
  }
}
